// LinearLoopLayer_21251498180729
// MI455X (gfx1250) — matrix operand order repaired and checked by reading, not compiled, not run on hardware
//
#include <hip/hip_runtime.h>

// out[M,N] = x[M,K] @ W[N,K]^T + b[N]   (M=8192, N=2048, K=2048, fp32)
// bf16x3 split-precision GEMM on v_wmma_f32_16x16x32_bf16 (CDNA5 / gfx1250).
// Block tile 256x128 (256 threads = 8 waves, 4x2), wave tile 64x64 (4x4 WMMA tiles).

typedef __attribute__((ext_vector_type(16))) __bf16 v16bf;
typedef __attribute__((ext_vector_type(8)))  float  v8f;

#define BM 256
#define BN 128
#define BK 32
#define LDK 40          // LDS row pitch in halves: 32 + 8 pad -> conflict-free b128 reads
#define THREADS 256

// Split a float4 into hi/lo bf16 (hi = RNE(f), lo = RNE(f - hi)) and store
// 4 packed halves to each LDS array. Native __bf16 casts -> v_cvt_pk_bf16_f32.
__device__ __forceinline__ void split_store(unsigned short* hiArr, unsigned short* loArr,
                                            int row, int kc, float4 v) {
  float f[4] = {v.x, v.y, v.z, v.w};
  unsigned hs[4], ls[4];
#pragma unroll
  for (int i = 0; i < 4; ++i) {
    __bf16 h = (__bf16)f[i];
    float hf = (float)h;
    __bf16 l = (__bf16)(f[i] - hf);
    unsigned short hu, lu;
    __builtin_memcpy(&hu, &h, 2);
    __builtin_memcpy(&lu, &l, 2);
    hs[i] = hu;
    ls[i] = lu;
  }
  uint2 hv = make_uint2(hs[0] | (hs[1] << 16), hs[2] | (hs[3] << 16));
  uint2 lv = make_uint2(ls[0] | (ls[1] << 16), ls[2] | (ls[3] << 16));
  *(uint2*)&hiArr[row * LDK + kc] = hv;
  *(uint2*)&loArr[row * LDK + kc] = lv;
}

union Frag { uint4 q[2]; v16bf v; };

// A fragment (16x32 bf16): lanes 0-15 hold K=0..7 (v0..3) and K=16..23 (v4..7);
// lanes 16-31 hold K=8..15 and K=24..31.
__device__ __forceinline__ v16bf load_a_frag(const unsigned short* s, int row, int lane) {
  const int kb = (lane >> 4) << 3;                 // 0 or 8
  const unsigned short* p = s + row * LDK;
  Frag f;
  f.q[0] = *(const uint4*)(p + kb);                // K = kb .. kb+7
  f.q[1] = *(const uint4*)(p + kb + 16);           // K = kb+16 .. kb+23
  return f.v;
}

// B fragment (32x16 bf16): lanes 0-15 (N=0..15) hold K=0..7 and K=16..23; lanes 16-31 hold K=8..15 and K=24..31.
__device__ __forceinline__ v16bf load_b_frag(const unsigned short* s, int row, int lane) {
  const int kb = (lane >> 4) << 3;                 // 0 or 8
  const unsigned short* p = s + row * LDK + kb;
  Frag f;
  f.q[0] = *(const uint4*)(p);                     // K = kb .. kb+7
  f.q[1] = *(const uint4*)(p + 16);                // K = kb+16 .. kb+23
  return f.v;
}

__global__ void __launch_bounds__(THREADS)
linear_wmma_bf16x3(const float* __restrict__ X, const float* __restrict__ Wm,
                   const float* __restrict__ bias, float* __restrict__ Out,
                   int M, int N, int K) {
  __shared__ __align__(16) unsigned short sAh[BM * LDK];   // 20 KB
  __shared__ __align__(16) unsigned short sAl[BM * LDK];   // 20 KB
  __shared__ __align__(16) unsigned short sBh[BN * LDK];   // 10 KB
  __shared__ __align__(16) unsigned short sBl[BN * LDK];   // 10 KB

  const int tid  = threadIdx.x;
  const int lane = tid & 31;
  const int wave = tid >> 5;
  const int wm   = wave & 3;     // 4 wave-rows -> 64 rows each (256 total)
  const int wn   = wave >> 2;    // 2 wave-cols -> 64 cols each (128 total)

  const int gm = blockIdx.y * BM;
  const int gn = blockIdx.x * BN;

  v8f acc[4][4];
#pragma unroll
  for (int mt = 0; mt < 4; ++mt)
#pragma unroll
    for (int nt = 0; nt < 4; ++nt)
      acc[mt][nt] = {};

  float4 xs[8], ws[4];

  // ---- prologue: global-load K-chunk 0 into registers ----
#pragma unroll
  for (int p = 0; p < 8; ++p) {
    int i = tid + p * THREADS;
    int row = i >> 3, c4 = (i & 7) << 2;
    xs[p] = *(const float4*)(X + (size_t)(gm + row) * K + c4);
  }
#pragma unroll
  for (int p = 0; p < 4; ++p) {
    int i = tid + p * THREADS;
    int row = i >> 3, c4 = (i & 7) << 2;
    ws[p] = *(const float4*)(Wm + (size_t)(gn + row) * K + c4);
  }

  const int KSTEPS = K / BK;
  for (int ks = 0; ks < KSTEPS; ++ks) {
    // stash current chunk (fp32 -> bf16 hi/lo) into LDS
#pragma unroll
    for (int p = 0; p < 8; ++p) {
      int i = tid + p * THREADS;
      int row = i >> 3, c4 = (i & 7) << 2;
      split_store(sAh, sAl, row, c4, xs[p]);
    }
#pragma unroll
    for (int p = 0; p < 4; ++p) {
      int i = tid + p * THREADS;
      int row = i >> 3, c4 = (i & 7) << 2;
      split_store(sBh, sBl, row, c4, ws[p]);
    }
    __syncthreads();

    // prefetch next chunk from global while WMMAs run
    if (ks + 1 < KSTEPS) {
      const int k0 = (ks + 1) * BK;
#pragma unroll
      for (int p = 0; p < 8; ++p) {
        int i = tid + p * THREADS;
        int row = i >> 3, c4 = (i & 7) << 2;
        xs[p] = *(const float4*)(X + (size_t)(gm + row) * K + k0 + c4);
      }
#pragma unroll
      for (int p = 0; p < 4; ++p) {
        int i = tid + p * THREADS;
        int row = i >> 3, c4 = (i & 7) << 2;
        ws[p] = *(const float4*)(Wm + (size_t)(gn + row) * K + k0 + c4);
      }
    }

    v16bf ah[4], al[4], bh[4], bl[4];
#pragma unroll
    for (int mt = 0; mt < 4; ++mt) {
      int row = wm * 64 + mt * 16 + (lane & 15);
      ah[mt] = load_a_frag(sAh, row, lane);
      al[mt] = load_a_frag(sAl, row, lane);
    }
#pragma unroll
    for (int nt = 0; nt < 4; ++nt) {
      int row = wn * 64 + nt * 16 + (lane & 15);
      bh[nt] = load_b_frag(sBh, row, lane);
      bl[nt] = load_b_frag(sBl, row, lane);
    }

#pragma unroll
    for (int mt = 0; mt < 4; ++mt) {
#pragma unroll
      for (int nt = 0; nt < 4; ++nt) {
        // x*w ~= xh*wh + xh*wl + xl*wh   (xl*wl ~ 2^-18 rel, negligible)
        acc[mt][nt] = __builtin_amdgcn_wmma_f32_16x16x32_bf16(
            false, ah[mt], false, bh[nt], (short)0, acc[mt][nt], false, false);
        acc[mt][nt] = __builtin_amdgcn_wmma_f32_16x16x32_bf16(
            false, ah[mt], false, bl[nt], (short)0, acc[mt][nt], false, false);
        acc[mt][nt] = __builtin_amdgcn_wmma_f32_16x16x32_bf16(
            false, al[mt], false, bh[nt], (short)0, acc[mt][nt], false, false);
      }
    }
    __syncthreads();
  }

  // ---- epilogue: bias add + store ----
  // C/D layout: lane -> N = lane&15; VGPR r -> M = r + 8*(lane>>4)
#pragma unroll
  for (int nt = 0; nt < 4; ++nt) {
    int col = gn + wn * 64 + nt * 16 + (lane & 15);
    float bv = bias[col];
#pragma unroll
    for (int mt = 0; mt < 4; ++mt) {
      int rbase = gm + wm * 64 + mt * 16 + ((lane >> 4) << 3);
#pragma unroll
      for (int r = 0; r < 8; ++r) {
        Out[(size_t)(rbase + r) * N + col] = acc[mt][nt][r] + bv;
      }
    }
  }
}

extern "C" void kernel_launch(void* const* d_in, const int* in_sizes, int n_in,
                              void* d_out, int out_size, void* d_ws, size_t ws_size,
                              hipStream_t stream) {
  const float* x = (const float*)d_in[0];
  const float* W = (const float*)d_in[1];
  const float* b = (const float*)d_in[2];
  float* out = (float*)d_out;

  const int Nn = in_sizes[2];           // OUT = 2048
  const int Kk = in_sizes[1] / Nn;      // IN  = 2048
  const int Mm = in_sizes[0] / Kk;      // B   = 8192

  dim3 grid(Nn / BN, Mm / BM);          // (16, 32)
  linear_wmma_bf16x3<<<grid, dim3(THREADS), 0, stream>>>(x, W, b, out, Mm, Nn, Kk);
}
